// JointNetwork_27479200760216
// MI455X (gfx1250) — compile-verified, matrix operand order checked by reading, not run on hardware
//
#include <hip/hip_runtime.h>
#include <hip/hip_bf16.h>

typedef __bf16 bf16_t;
typedef __attribute__((ext_vector_type(8)))  __bf16 v8bf;
typedef __attribute__((ext_vector_type(16))) __bf16 v16bf;
typedef __attribute__((ext_vector_type(8)))  float  v8f;

#define B_DIM 4
#define T_DIM 256
#define U_DIM 64
#define JD    512      // ENC_DIM == PRED_DIM == JOINT_DIM
#define VOC   1024
#define LDA   528      // padded LDS row stride (bf16 elems): 528*2B = 1056B, breaks bank conflicts

// ---------------------------------------------------------------------------
// fast tanh: tanh(x) = 1 - 2/(exp2(x*2*log2e)+1); saturates correctly at +/-inf
__device__ __forceinline__ float fast_tanh(float x) {
    float e = __builtin_amdgcn_exp2f(x * 2.885390081777927f); // 2*log2(e)
    return 1.0f - 2.0f * __builtin_amdgcn_rcpf(e + 1.0f);
}

// build v16bf fragment from two contiguous 8-element (16B) chunks
__device__ __forceinline__ v16bf frag16(const bf16_t* p) {
    v8bf lo = *(const v8bf*)(p);
    v8bf hi = *(const v8bf*)(p + 16);
    return __builtin_shufflevector(lo, hi, 0,1,2,3,4,5,6,7,8,9,10,11,12,13,14,15);
}

// ---------------------------------------------------------------------------
// One-time weight transpose/convert:
//   WtJ   [v][j] = bf16(W_joint[j][v])                (1024 x 512)
//   WtE_hi/lo [n][k] = split(W_enc[k][n])             (512 x 512)
//   WtP_hi/lo [n][k] = split(W_pred[k][n])            (512 x 512)
__global__ __launch_bounds__(256) void prep_kernel(
    const float* __restrict__ Wj, const float* __restrict__ We, const float* __restrict__ Wp,
    bf16_t* __restrict__ WtJ,
    bf16_t* __restrict__ WtE_hi, bf16_t* __restrict__ WtE_lo,
    bf16_t* __restrict__ WtP_hi, bf16_t* __restrict__ WtP_lo)
{
    int i = blockIdx.x * blockDim.x + threadIdx.x;
    if (i < 512 * 1024) {
        int v = i >> 9, j = i & 511;
        WtJ[i] = (bf16_t)Wj[j * VOC + v];
    } else if (i < 512 * 1024 + 512 * 512) {
        int r = i - 512 * 1024;
        int n = r >> 9, k = r & 511;
        float x = We[k * JD + n];
        bf16_t h = (bf16_t)x;
        WtE_hi[r] = h;
        WtE_lo[r] = (bf16_t)(x - (float)h);
    } else {
        int r = i - (512 * 1024 + 512 * 512);
        int n = r >> 9, k = r & 511;
        float x = Wp[k * JD + n];
        bf16_t h = (bf16_t)x;
        WtP_hi[r] = h;
        WtP_lo[r] = (bf16_t)(x - (float)h);
    }
}

// ---------------------------------------------------------------------------
// Projection GEMM: Y(M x 512) = X(M x 512) * W(512 x 512) + bias
// W given pre-transposed as bf16 hi/lo (N-major). bf16 split-precision:
// X = Xhi + Xlo, W = Whi + Wlo ; Y ~= Xhi*Whi + Xlo*Whi + Xhi*Wlo (3 WMMAs/K-step)
__global__ __launch_bounds__(256) void proj_kernel(
    const float* __restrict__ X,
    const bf16_t* __restrict__ Wt_hi, const bf16_t* __restrict__ Wt_lo,
    const float* __restrict__ bias,
    float* __restrict__ Y)
{
    __shared__ bf16_t Ahi[16 * LDA];
    __shared__ bf16_t Alo[16 * LDA];

    const int m0  = blockIdx.x * 16;
    const int tid = threadIdx.x;

    // cooperative load + hi/lo split of the 16 x 512 input strip
    {
        int r = tid >> 4, c16 = tid & 15;
        const float4* px = (const float4*)(X + (size_t)(m0 + r) * JD);
        #pragma unroll
        for (int ch = 0; ch < 8; ++ch) {
            int i4 = c16 + ch * 16;          // float4 index within row
            float4 x = px[i4];
            bf16_t* dh = &Ahi[r * LDA + i4 * 4];
            bf16_t* dl = &Alo[r * LDA + i4 * 4];
            float xs[4] = {x.x, x.y, x.z, x.w};
            #pragma unroll
            for (int j = 0; j < 4; ++j) {
                bf16_t h = (bf16_t)xs[j];
                dh[j] = h;
                dl[j] = (bf16_t)(xs[j] - (float)h);
            }
        }
    }
    __syncthreads();

    const int wave = tid >> 5;
    const int lane = tid & 31;
    const int g    = lane >> 4;     // lane group 0/1
    const int lr   = lane & 15;

    v8f c[4] = {};                  // 4 n-tiles per wave: 8 waves * 4 * 16 = 512 cols

    for (int ks = 0; ks < 16; ++ks) {
        const int k0 = ks * 32 + g * 8;   // chunk0; chunk1 at +16
        v16bf ahi = frag16(&Ahi[lr * LDA + k0]);
        v16bf alo = frag16(&Alo[lr * LDA + k0]);
        #pragma unroll
        for (int nt = 0; nt < 4; ++nt) {
            int n = (wave * 4 + nt) * 16 + lr;       // B column for this lane
            v16bf bhi = frag16(Wt_hi + (size_t)n * JD + k0);
            v16bf blo = frag16(Wt_lo + (size_t)n * JD + k0);
            c[nt] = __builtin_amdgcn_wmma_f32_16x16x32_bf16(false, ahi, false, bhi, (short)0, c[nt], false, false);
            c[nt] = __builtin_amdgcn_wmma_f32_16x16x32_bf16(false, alo, false, bhi, (short)0, c[nt], false, false);
            c[nt] = __builtin_amdgcn_wmma_f32_16x16x32_bf16(false, ahi, false, blo, (short)0, c[nt], false, false);
        }
    }

    // store: C layout -> lane lr = column, VGPR v = row (v + 8*g)
    #pragma unroll
    for (int nt = 0; nt < 4; ++nt) {
        int n = (wave * 4 + nt) * 16 + lr;
        float bs = bias[n];
        float* op = Y + (size_t)(m0 + g * 8) * JD + n;
        #pragma unroll
        for (int v = 0; v < 8; ++v)
            op[(size_t)v * JD] = c[nt][v] + bs;
    }
}

// ---------------------------------------------------------------------------
// Joint GEMM: out(65536 x 1024) = tanh(encp[bt] + predp[bu]) (bf16) * WtJ + b_joint
//
// Block = one (b,t): M_blk = 64 rows (all 64 u), N_blk = 512 cols (grid.x = 2).
// The 64x512 tanh strip is built once in LDS (67.6 KB). Each of 8 waves owns a
// 64x64 output tile = 4 M-tiles x 4 N-tiles: per K-step, 4 A-frags (LDS) and
// 4 B-frags (global) feed 16 WMMAs -> each B line fetched by exactly one wave,
// L2 B-read traffic drops 4x vs a 16-row block (4 GB -> 1 GB total).
__global__ __launch_bounds__(256) void joint_kernel(
    const float* __restrict__ encp,     // (B*T, 512)
    const float* __restrict__ predp,    // (B*U, 512)
    const bf16_t* __restrict__ WtJ,     // (1024, 512) bf16, vocab-major
    const float* __restrict__ bj,       // (1024)
    float* __restrict__ out)            // (B*T*U, 1024)
{
    __shared__ bf16_t Asm[64 * LDA];

    const int bt  = blockIdx.y;              // 0..1023 : (b,t) pair
    const int b   = bt >> 8;                 // bt / T_DIM
    const int nb0 = blockIdx.x * 512;        // N-block base (0 or 512)
    const size_t m0 = (size_t)bt * U_DIM;    // global M row base

    const float* erow  = encp  + (size_t)bt * JD;
    const float* prow0 = predp + (size_t)(b * U_DIM) * JD;

    const int tid = threadIdx.x;

    // cooperative: A[u][k] = tanh(enc[k] + pred[u][k]) -> bf16 LDS strip (64 x 512)
    {
        int r16 = tid >> 4, c16 = tid & 15;
        const float4* pe = (const float4*)erow;
        #pragma unroll
        for (int rr = 0; rr < 4; ++rr) {
            int r = rr * 16 + r16;           // row (u) 0..63
            const float4* pp = (const float4*)(prow0 + (size_t)r * JD);
            #pragma unroll
            for (int ch = 0; ch < 8; ++ch) {
                int i4 = c16 + ch * 16;      // float4 index within row
                float4 e = pe[i4];
                float4 p = pp[i4];
                bf16_t* dst = &Asm[r * LDA + i4 * 4];
                dst[0] = (bf16_t)fast_tanh(e.x + p.x);
                dst[1] = (bf16_t)fast_tanh(e.y + p.y);
                dst[2] = (bf16_t)fast_tanh(e.z + p.z);
                dst[3] = (bf16_t)fast_tanh(e.w + p.w);
            }
        }
    }
    __syncthreads();

    const int wave = tid >> 5;
    const int lane = tid & 31;
    const int g    = lane >> 4;
    const int lr   = lane & 15;
    const int ncol0 = nb0 + wave * 64;       // this wave's 64-column slab

    v8f c[4][4] = {};                        // [m-tile][n-tile]

    for (int ks = 0; ks < 16; ++ks) {
        const int k0 = ks * 32 + g * 8;      // chunk0; chunk1 at +16
        v16bf a[4];
        #pragma unroll
        for (int mt = 0; mt < 4; ++mt)
            a[mt] = frag16(&Asm[(mt * 16 + lr) * LDA + k0]);
        #pragma unroll
        for (int nt = 0; nt < 4; ++nt) {
            int n = ncol0 + nt * 16 + lr;    // B column for this lane
            v16bf bf = frag16(WtJ + (size_t)n * JD + k0);
            #pragma unroll
            for (int mt = 0; mt < 4; ++mt)
                c[mt][nt] = __builtin_amdgcn_wmma_f32_16x16x32_bf16(false, a[mt], false, bf, (short)0, c[mt][nt], false, false);
        }
    }

    // store: C layout -> lane lr = column, VGPR v = row (v + 8*g); fused bias
    #pragma unroll
    for (int nt = 0; nt < 4; ++nt) {
        int n = ncol0 + nt * 16 + lr;
        float bs = bj[n];
        #pragma unroll
        for (int mt = 0; mt < 4; ++mt) {
            float* op = out + (m0 + mt * 16 + g * 8) * VOC + n;
            #pragma unroll
            for (int v = 0; v < 8; ++v)
                op[(size_t)v * VOC] = c[mt][nt][v] + bs;
        }
    }
}

// ---------------------------------------------------------------------------
extern "C" void kernel_launch(void* const* d_in, const int* in_sizes, int n_in,
                              void* d_out, int out_size, void* d_ws, size_t ws_size,
                              hipStream_t stream) {
    (void)in_sizes; (void)n_in; (void)out_size; (void)ws_size;
    const float* enc  = (const float*)d_in[0];   // (4,256,512)
    const float* pred = (const float*)d_in[1];   // (4,64,512)
    const float* We   = (const float*)d_in[2];   // (512,512)
    const float* be   = (const float*)d_in[3];   // (512)
    const float* Wp   = (const float*)d_in[4];   // (512,512)
    const float* bp   = (const float*)d_in[5];   // (512)
    const float* Wj   = (const float*)d_in[6];   // (512,1024)
    const float* bj   = (const float*)d_in[7];   // (1024)
    float* out = (float*)d_out;

    char* ws = (char*)d_ws;
    float*  encp    = (float*) (ws + 0);          // 1024*512*4 = 2097152
    float*  predp   = (float*) (ws + 2097152);    //  256*512*4 =  524288
    bf16_t* WtJ     = (bf16_t*)(ws + 2621440);    // 1024*512*2 = 1048576
    bf16_t* WtE_hi  = (bf16_t*)(ws + 3670016);    //  512*512*2 =  524288
    bf16_t* WtE_lo  = (bf16_t*)(ws + 4194304);
    bf16_t* WtP_hi  = (bf16_t*)(ws + 4718592);
    bf16_t* WtP_lo  = (bf16_t*)(ws + 5242880);    // total 5767168 bytes

    // 1) weight transpose/convert (1,048,576 work items)
    prep_kernel<<<4096, 256, 0, stream>>>(Wj, We, Wp, WtJ, WtE_hi, WtE_lo, WtP_hi, WtP_lo);

    // 2) projections (split-precision bf16 WMMA, f32 accumulate)
    proj_kernel<<<64, 256, 0, stream>>>(enc,  WtE_hi, WtE_lo, be, encp);   // 1024 rows
    proj_kernel<<<16, 256, 0, stream>>>(pred, WtP_hi, WtP_lo, bp, predp);  //  256 rows

    // 3) fused tanh + vocab GEMM (bf16 WMMA, f32 accumulate, fused bias)
    //    grid: (N-blocks = 2, (b,t) blocks = 1024)
    joint_kernel<<<dim3(2, 1024), 256, 0, stream>>>(encp, predp, WtJ, bj, out);
}
